// word_sen_21328807592158
// MI455X (gfx1250) — matrix operand order repaired and checked by reading, not compiled, not run on hardware
//
#include <hip/hip_runtime.h>

#define B_  4
#define S_  2048
#define H_  768
#define NH_ 12
#define DH_ 64

typedef __attribute__((ext_vector_type(8)))  float    f32x8;
typedef __attribute__((ext_vector_type(8)))  _Float16 f16x8;
typedef __attribute__((ext_vector_type(16))) _Float16 f16x16;

// D = A*B + C, 16x16x32 f16 -> f32
#define WMMA_F16(A, Bm, C) \
  __builtin_amdgcn_wmma_f32_16x16x32_f16(false, (A), false, (Bm), (short)0, (C), false, false)

static __device__ __forceinline__ f16x16 pack_f16(f16x8 a, f16x8 b) {
  f16x16 r;
#pragma unroll
  for (int i = 0; i < 8; ++i) { r[i] = a[i]; r[8 + i] = b[i]; }
  return r;
}

// XOR-tree reduction across the 16-lane half the caller's row lives in.
static __device__ __forceinline__ float rowred_max(float v) {
  v = fmaxf(v, __int_as_float(__builtin_amdgcn_ds_swizzle(__float_as_int(v), 0x041F)));
  v = fmaxf(v, __int_as_float(__builtin_amdgcn_ds_swizzle(__float_as_int(v), 0x081F)));
  v = fmaxf(v, __int_as_float(__builtin_amdgcn_ds_swizzle(__float_as_int(v), 0x101F)));
  v = fmaxf(v, __int_as_float(__builtin_amdgcn_ds_swizzle(__float_as_int(v), 0x201F)));
  return v;
}
static __device__ __forceinline__ float rowred_sum(float v) {
  v += __int_as_float(__builtin_amdgcn_ds_swizzle(__float_as_int(v), 0x041F));
  v += __int_as_float(__builtin_amdgcn_ds_swizzle(__float_as_int(v), 0x081F));
  v += __int_as_float(__builtin_amdgcn_ds_swizzle(__float_as_int(v), 0x101F));
  v += __int_as_float(__builtin_amdgcn_ds_swizzle(__float_as_int(v), 0x201F));
  return v;
}

// ---------------------------------------------------------------------------
// One-shot fp32 -> f16 converter (n must be a multiple of 8).
// ---------------------------------------------------------------------------
__global__ __launch_bounds__(256) void cvt_f16_kernel(
    const float* __restrict__ in, _Float16* __restrict__ outp, int n)
{
  const int i = (blockIdx.x * 256 + threadIdx.x) * 8;
  if (i < n) {
    f32x8 v = *(const f32x8*)(in + i);
    f16x8 o;
#pragma unroll
    for (int j = 0; j < 8; ++j) o[j] = (_Float16)v[j];
    *(f16x8*)(outp + i) = o;
  }
}

// ---------------------------------------------------------------------------
// Projection GEMM (all-f16): out[m,n] = sum_k X[m,k]*W[n,k] + bias[n]
// 256 threads = 8 waves; tile 64(M) x 128(N); each wave: four 16x16 WMMA
// accumulators.  K loop FULLY UNROLLED (24 steps): no back-edge phis => no
// register copies; all addresses are base+immediate => no address VALU; the
// scheduler pipelines loads across WMMAs up to register pressure.
// storeT==0: [B,NH,S,DH] (Q/K/SQ/SK).  storeT==1: [B,NH,DH,S] (V^T).
// ---------------------------------------------------------------------------
__global__ __launch_bounds__(256) void proj_gemm_kernel(
    const _Float16* __restrict__ X, const _Float16* __restrict__ W,
    const float* __restrict__ bias, _Float16* __restrict__ out, int storeT)
{
  const int lane = threadIdx.x & 31;
  const int wave = threadIdx.x >> 5;
  const int mw = wave & 3, nw = wave >> 2;
  const int m_base = blockIdx.y * 64 + mw * 16;
  const int n_base = blockIdx.x * 128 + nw * 64;

  const int l15  = lane & 15;
  const int arow = m_base + l15;
  const int alo  = (lane < 16) ? 0 : 16;   // A-frag K chunk start
  const int bko  = (lane < 16) ? 0 : 16;   // B-frag K half

  const _Float16* ap  = X + (size_t)arow * H_ + alo;
  const _Float16* wp0 = W + (size_t)(n_base + l15) * H_ + bko;

  f32x8 acc[4] = {};

#pragma unroll
  for (int k = 0; k < H_; k += 32) {
    f16x16 af = pack_f16(*(const f16x8*)(ap + k), *(const f16x8*)(ap + k + 8));
#pragma unroll
    for (int s = 0; s < 4; ++s) {
      f16x16 bf = *(const f16x16*)(wp0 + (size_t)s * 16 * H_ + k);
      acc[s] = WMMA_F16(af, bf, acc[s]);
    }
  }

#pragma unroll
  for (int s = 0; s < 4; ++s) {
    const int n = n_base + s * 16 + l15;
    const float bn = bias[n];
    const int h = n >> 6, d = n & 63;
#pragma unroll
    for (int r = 0; r < 8; ++r) {
      const int mrow = m_base + ((lane < 16) ? r : 8 + r);
      const int bb = mrow >> 11, ss = mrow & (S_ - 1);
      const size_t addr = storeT
          ? ((((size_t)bb * NH_ + h) * DH_ + d) * S_ + ss)
          : ((((size_t)bb * NH_ + h) * S_ + ss) * DH_ + d);
      out[addr] = (_Float16)(acc[s][r] + bn);
    }
  }
}

// ---------------------------------------------------------------------------
// Fused dual-score flash attention.
// Grid: (S/64, B*NH). Block = 128 threads = 4 waves; wave owns 16 query rows.
// All 12 fragment loads for a kk-tile are issued up front in distinct
// registers; V fragments arrive under the shadow of 8 score-WMMAs + softmax.
// ---------------------------------------------------------------------------
__global__ __launch_bounds__(128) void dual_flash_attn_kernel(
    const _Float16* __restrict__ Q,  const _Float16* __restrict__ K,
    const _Float16* __restrict__ Vt, const _Float16* __restrict__ SQ,
    const _Float16* __restrict__ SK, const int* __restrict__ mask,
    float* __restrict__ out)
{
  __shared__ __align__(16) _Float16 pbuf[4][16][32];

  const int lane = threadIdx.x & 31;
  const int wave = threadIdx.x >> 5;
  const int bh = blockIdx.y;
  const int b = bh / NH_, h = bh % NH_;
  const int q0 = blockIdx.x * 64 + wave * 16;

  const _Float16* Qb  = Q  + (size_t)bh * S_ * DH_;
  const _Float16* Kb  = K  + (size_t)bh * S_ * DH_;
  const _Float16* Vb  = Vt + (size_t)bh * DH_ * S_;
  const _Float16* SQb = SQ + (size_t)bh * S_ * DH_;
  const _Float16* SKb = SK + (size_t)bh * S_ * DH_;
  const int* mb = mask + b * S_;

  const int l15 = lane & 15;
  const int alo = (lane < 16) ? 0 : 16;
  const int bko = (lane < 16) ? 0 : 16;
  const int arow = q0 + l15;

  // Q / SQ A-fragments for the whole row block (d = 0..31, 32..63)
  f16x16 qa[2], sqa[2];
#pragma unroll
  for (int t = 0; t < 2; ++t) {
    const _Float16* p  = Qb  + (size_t)arow * DH_ + t * 32 + alo;
    const _Float16* ps = SQb + (size_t)arow * DH_ + t * 32 + alo;
    qa[t]  = pack_f16(*(const f16x8*)(p),  *(const f16x8*)(p  + 8));
    sqa[t] = pack_f16(*(const f16x8*)(ps), *(const f16x8*)(ps + 8));
  }

  f32x8 acco[4] = {};
  float row_m[8], row_l[8];
#pragma unroll
  for (int r = 0; r < 8; ++r) { row_m[r] = -3.0e38f; row_l[r] = 0.0f; }

  const float inv = 0.08838834764831845f;  // 1/sqrt(2*DH)

#pragma unroll 1
  for (int kk0 = 0; kk0 < S_; kk0 += 32) {
    // ---- issue ALL loads for this kk-tile up front (distinct registers) ----
    const int col0 = kk0 + l15;
    const _Float16* kp0  = Kb  + (size_t)col0 * DH_ + bko;
    const _Float16* kp1  = kp0 + 16 * DH_;
    const _Float16* skp0 = SKb + (size_t)col0 * DH_ + bko;
    const _Float16* skp1 = skp0 + 16 * DH_;

    f16x16 kb00  = *(const f16x16*)(kp0);
    f16x16 kb01  = *(const f16x16*)(kp0 + 32);
    f16x16 kb10  = *(const f16x16*)(kp1);
    f16x16 kb11  = *(const f16x16*)(kp1 + 32);
    f16x16 skb00 = *(const f16x16*)(skp0);
    f16x16 skb01 = *(const f16x16*)(skp0 + 32);
    f16x16 skb10 = *(const f16x16*)(skp1);
    f16x16 skb11 = *(const f16x16*)(skp1 + 32);

    f16x16 vf[4];
#pragma unroll
    for (int t = 0; t < 4; ++t)
      vf[t] = *(const f16x16*)(Vb + (size_t)(t * 16 + l15) * S_ + kk0 + bko);

    const int mv0 = mb[col0];
    const int mv1 = mb[col0 + 16];

    // prefetch next tile's K/SK/V fragments into the near cache
    if (kk0 + 32 < S_) {
      __builtin_prefetch(kp0 + 32 * DH_, 0, 0);
      __builtin_prefetch(skp0 + 32 * DH_, 0, 0);
      __builtin_prefetch(Vb + (size_t)l15 * S_ + kk0 + 32 + bko, 0, 0);
    }

    // ---- dual scores: 8 WMMAs ----
    f32x8 sc[2];
    {
      f32x8 s0 = {}, s1 = {};
      s0 = WMMA_F16(qa[0],  kb00,  s0);
      s0 = WMMA_F16(qa[1],  kb01,  s0);
      s0 = WMMA_F16(sqa[0], skb00, s0);
      s0 = WMMA_F16(sqa[1], skb01, s0);
      s1 = WMMA_F16(qa[0],  kb10,  s1);
      s1 = WMMA_F16(qa[1],  kb11,  s1);
      s1 = WMMA_F16(sqa[0], skb10, s1);
      s1 = WMMA_F16(sqa[1], skb11, s1);
      const float pen0 = mv0 ? 0.0f : 10000.0f;
      const float pen1 = mv1 ? 0.0f : 10000.0f;
#pragma unroll
      for (int r = 0; r < 8; ++r) {
        s0[r] = s0[r] * inv - pen0;
        s1[r] = s1[r] * inv - pen1;
      }
      sc[0] = s0; sc[1] = s1;
    }

    // ---- online softmax over the 32 new columns ----
    float alpha[8];
#pragma unroll
    for (int r = 0; r < 8; ++r) {
      float mx = rowred_max(fmaxf(sc[0][r], sc[1][r]));
      float nm = fmaxf(row_m[r], mx);
      float a  = __expf(row_m[r] - nm);
      float p0 = __expf(sc[0][r] - nm);
      float p1 = __expf(sc[1][r] - nm);
      sc[0][r] = p0; sc[1][r] = p1;
      row_l[r] = row_l[r] * a + rowred_sum(p0 + p1);
      row_m[r] = nm;
      alpha[r] = a;
    }
#pragma unroll
    for (int t = 0; t < 4; ++t)
#pragma unroll
      for (int r = 0; r < 8; ++r) acco[t][r] *= alpha[r];

    // ---- transpose P (C-layout) -> A-layout through per-wave LDS tile ----
    const int prow = (lane < 16) ? 0 : 8;
#pragma unroll
    for (int r = 0; r < 8; ++r) {
      pbuf[wave][prow + r][l15]      = (_Float16)sc[0][r];
      pbuf[wave][prow + r][16 + l15] = (_Float16)sc[1][r];
    }
    __asm__ volatile("s_wait_dscnt 0" ::: "memory");

    const int c0 = (lane < 16) ? 0 : 16;
    f16x16 pa = pack_f16(*(const f16x8*)&pbuf[wave][l15][c0],
                         *(const f16x8*)&pbuf[wave][l15][c0 + 8]);

    // ---- O += P * V (V fragments already resident) ----
#pragma unroll
    for (int t = 0; t < 4; ++t)
      acco[t] = WMMA_F16(pa, vf[t], acco[t]);
  }

  // normalize + store [B,S,H] with head interleave
#pragma unroll
  for (int r = 0; r < 8; ++r) row_l[r] = 1.0f / row_l[r];
#pragma unroll
  for (int t = 0; t < 4; ++t) {
    const int d = t * 16 + l15;
#pragma unroll
    for (int r = 0; r < 8; ++r) {
      const int qrow = q0 + ((lane < 16) ? r : 8 + r);
      out[((size_t)b * S_ + qrow) * H_ + h * DH_ + d] = acco[t][r] * row_l[r];
    }
  }
}

extern "C" void kernel_launch(void* const* d_in, const int* in_sizes, int n_in,
                              void* d_out, int out_size, void* d_ws, size_t ws_size,
                              hipStream_t stream)
{
  (void)in_sizes; (void)n_in; (void)out_size; (void)ws_size;
  const float* word = (const float*)d_in[0];
  const float* sen  = (const float*)d_in[1];
  const int*   mask = (const int*)d_in[2];
  const float* wW[5] = {(const float*)d_in[3], (const float*)d_in[5], (const float*)d_in[7],
                        (const float*)d_in[9], (const float*)d_in[11]};
  const float* wB[5] = {(const float*)d_in[4], (const float*)d_in[6], (const float*)d_in[8],
                        (const float*)d_in[10], (const float*)d_in[12]};

  _Float16* ws = (_Float16*)d_ws;
  const size_t per = (size_t)B_ * S_ * H_;      // 6.29M f16 elems
  const size_t wsz = (size_t)H_ * H_;           // 589,824 f16 elems
  _Float16* Qp    = ws + 0 * per;               // [B,NH,S,DH]
  _Float16* Kp    = ws + 1 * per;               // [B,NH,S,DH]
  _Float16* Vtp   = ws + 2 * per;               // [B,NH,DH,S]
  _Float16* SQp   = ws + 3 * per;               // [B,NH,S,DH]
  _Float16* SKp   = ws + 4 * per;               // [B,NH,S,DH]
  _Float16* wordh = ws + 5 * per;               // [B*S, H] f16
  _Float16* senh  = ws + 6 * per;               // [B*S, H] f16
  _Float16* Wh[5];
  for (int i = 0; i < 5; ++i) Wh[i] = ws + 7 * per + (size_t)i * wsz;

  // stage 0: one-shot fp32 -> f16 conversions (activations + weights)
  {
    const int nact = (int)per;
    dim3 ga((nact / 8 + 255) / 256), ba(256);
    cvt_f16_kernel<<<ga, ba, 0, stream>>>(word, wordh, nact);
    cvt_f16_kernel<<<ga, ba, 0, stream>>>(sen,  senh,  nact);
    const int nw = (int)wsz;
    dim3 gw((nw / 8 + 255) / 256);
    for (int i = 0; i < 5; ++i)
      cvt_f16_kernel<<<gw, ba, 0, stream>>>(wW[i], Wh[i], nw);
  }

  // stage 1: five projection GEMMs (fully-unrolled f16 WMMA K loops)
  dim3 gA(H_ / 128, (B_ * S_) / 64);
  dim3 bA(256);
  proj_gemm_kernel<<<gA, bA, 0, stream>>>(wordh, Wh[0], wB[0], Qp, 0);
  proj_gemm_kernel<<<gA, bA, 0, stream>>>(wordh, Wh[1], wB[1], Kp, 0);
  proj_gemm_kernel<<<gA, bA, 0, stream>>>(wordh, Wh[2], wB[2], Vtp, 1);
  proj_gemm_kernel<<<gA, bA, 0, stream>>>(senh,  Wh[3], wB[3], SQp, 0);
  proj_gemm_kernel<<<gA, bA, 0, stream>>>(senh,  Wh[4], wB[4], SKp, 0);

  // stage 2: fused dual-score flash attention
  dim3 gB(S_ / 64, B_ * NH_);
  dim3 bB(128);
  dual_flash_attn_kernel<<<gB, bB, 0, stream>>>(Qp, Kp, Vtp, SQp, SKp, mask,
                                                (float*)d_out);
}
